// SelfAttention_1022202217382
// MI455X (gfx1250) — matrix operand order repaired and checked by reading, not compiled, not run on hardware
//
#include <hip/hip_runtime.h>

// MI455X / gfx1250, wave32 only.
typedef __attribute__((ext_vector_type(2)))  float        v2f;
typedef __attribute__((ext_vector_type(8)))  float        v8f;
typedef __attribute__((ext_vector_type(16))) __bf16       v16bf;
typedef __attribute__((ext_vector_type(8)))  __bf16       v8bf;
typedef __attribute__((ext_vector_type(4)))  unsigned int v4u;
typedef __attribute__((ext_vector_type(8)))  int          v8i;
typedef __attribute__((ext_vector_type(4)))  int          v4i;

#define N_B   8
#define C_DIM 256
#define S_DIM 4096

// A-fragment (16x32 bf16, M x K): per lane, 16 consecutive K [base..base+15]
static __device__ __forceinline__ v16bf load_frag_a(const __bf16* p) {
  v8bf lo = *(const v8bf*)p;
  v8bf hi = *(const v8bf*)(p + 8);
  v16bf r;
#pragma unroll
  for (int i = 0; i < 8; ++i) { r[i] = lo[i]; r[i + 8] = hi[i]; }
  return r;
}

// B-fragment (32x16 bf16, K x N): per lane, 16 consecutive K of one column
static __device__ __forceinline__ v16bf load_frag_b(const __bf16* p) {
  v8bf lo = *(const v8bf*)p;
  v8bf hi = *(const v8bf*)(p + 8);
  v16bf r;
#pragma unroll
  for (int i = 0; i < 8; ++i) { r[i] = lo[i]; r[i + 8] = hi[i]; }
  return r;
}

// ---------------------------------------------------------------------------
// TDM: 2-D tile DMA (global -> LDS), bf16 elements (data_size = 2 bytes).
// D# per CDNA5 ISA ch.8: group0 {count, lds_addr, global_addr, type=2},
// group1 {data_size, tensor dims, tile dims, dim0 stride}. Groups 2/3 zero.
// ---------------------------------------------------------------------------
static __device__ __forceinline__ void tdm_load_2d_bf16(
    unsigned int lds_byte_off, const void* gaddr,
    unsigned int tile_d0, unsigned int tile_d1,
    unsigned int tensor_d0, unsigned int tensor_d1,
    unsigned long long stride0_elems)
{
  unsigned long long ga = (unsigned long long)(uintptr_t)gaddr;
  v4u g0;
  g0[0] = 1u;                                   // count=1 (valid), user mode
  g0[1] = lds_byte_off;                         // LDS byte address
  g0[2] = (unsigned int)ga;                     // global_addr[31:0]
  g0[3] = (unsigned int)((ga >> 32) & 0x01FFFFFFull) | (2u << 30); // addr[56:32], type=2
  v8i g1;
  g1[0] = (int)(1u << 16);                      // workgroup_mask=0, data_size=1 (2B)
  g1[1] = (int)((tensor_d0 & 0xFFFFu) << 16);   // tensor_dim0[15:0]
  g1[2] = (int)((tensor_d0 >> 16) | ((tensor_d1 & 0xFFFFu) << 16));
  g1[3] = (int)((tensor_d1 >> 16) | (tile_d0 << 16));   // tile_dim0 @ [127:112]
  g1[4] = (int)tile_d1;                         // tile_dim1, tile_dim2=0
  g1[5] = (int)(unsigned int)(stride0_elems & 0xFFFFFFFFull);
  g1[6] = (int)(unsigned int)((stride0_elems >> 32) & 0xFFFFull); // dim1_stride=0
  g1[7] = 0;
  v4i z4 = {0, 0, 0, 0};
#if __has_include(<hip/amd_detail/amd_gfx1250_TDM.h>)
  v8i z8 = {0, 0, 0, 0, 0, 0, 0, 0};
  __builtin_amdgcn_tensor_load_to_lds(g0, g1, z4, z4, z8, 0);
#else
  __builtin_amdgcn_tensor_load_to_lds(g0, g1, z4, z4, 0);
#endif
}

// ---------------------------------------------------------------------------
// Kernel 1: projections.  q = Wq x, k = Wk x (bf16, (n,S,C)),
// v = Wv x (bf16, (n,C,S)), out = Wl x (f32, (n,C,S)).
// fp32 WMMA 16x16x4, K = C = 256. Wave -> (matrix, o-half); B fragment is
// loaded once per K-step and amortized over 8 o-tiles (8 accumulators).
// ---------------------------------------------------------------------------
__global__ __launch_bounds__(256) void proj_kernel(
    const float* __restrict__ xs,
    const float* __restrict__ Wq, const float* __restrict__ Wk,
    const float* __restrict__ Wv, const float* __restrict__ Wl,
    __bf16* __restrict__ qo, __bf16* __restrict__ ko,
    __bf16* __restrict__ vo, float* __restrict__ out)
{
  const int n    = blockIdx.y;
  const int s0   = blockIdx.x * 16;
  const int wave = threadIdx.x >> 5;
  const int lane = threadIdx.x & 31;
  const int half = lane >> 4;
  const int col  = lane & 15;
  const int mat  = wave >> 1;     // 0=q 1=k 2=v 3=l
  const int oh   = wave & 1;      // output-channel half

  const float* W = (mat == 0) ? Wq : (mat == 1) ? Wk : (mat == 2) ? Wv : Wl;
  const float* x = xs + (size_t)n * C_DIM * S_DIM;

  v8f acc[8];
#pragma unroll
  for (int ot = 0; ot < 8; ++ot)
#pragma unroll
    for (int r = 0; r < 8; ++r) acc[ot][r] = 0.0f;

  const float* bcol = x + (size_t)(2 * half) * S_DIM + s0 + col;
  const float* abase = W + (size_t)(oh * 128 + col) * C_DIM + 2 * half;
#pragma unroll 2
  for (int c0 = 0; c0 < C_DIM; c0 += 4) {
    v2f b;
    b.x = bcol[(size_t)c0 * S_DIM];
    b.y = bcol[(size_t)(c0 + 1) * S_DIM];
#pragma unroll
    for (int ot = 0; ot < 8; ++ot) {
      v2f a = *(const v2f*)(abase + (size_t)(ot * 16) * C_DIM + c0);
      acc[ot] = __builtin_amdgcn_wmma_f32_16x16x4_f32(
          false, a, false, b, (short)0, acc[ot], false, false);
    }
  }

#pragma unroll
  for (int ot = 0; ot < 8; ++ot) {
    const int o0 = oh * 128 + ot * 16;
    if (mat <= 1) {
      __bf16* dst = (mat == 0 ? qo : ko) +
                    ((size_t)n * S_DIM + s0 + col) * C_DIM + o0 + 8 * half;
      v8bf o8;
#pragma unroll
      for (int r = 0; r < 8; ++r) o8[r] = (__bf16)acc[ot][r];
      *(v8bf*)dst = o8;
    } else if (mat == 2) {
#pragma unroll
      for (int r = 0; r < 8; ++r)
        vo[((size_t)n * C_DIM + o0 + 8 * half + r) * S_DIM + s0 + col] =
            (__bf16)acc[ot][r];
    } else {
#pragma unroll
      for (int r = 0; r < 8; ++r)
        out[((size_t)n * C_DIM + o0 + 8 * half + r) * S_DIM + s0 + col] =
            acc[ot][r];
    }
  }
}

// ---------------------------------------------------------------------------
// Kernel 2 (pass A): per-key log-sum-exp over the query axis.
// e[t,s] = sum_c k[c,t] q[c,s]; L[t] = m_t + log(sum_s exp(e[t,s]-m_t)).
// ---------------------------------------------------------------------------
__global__ __launch_bounds__(256) void attn_stats_kernel(
    const __bf16* __restrict__ kin, const __bf16* __restrict__ qin,
    float* __restrict__ Lout)
{
  const int n    = blockIdx.y;
  const int wave = threadIdx.x >> 5;
  const int lane = threadIdx.x & 31;
  const int half = lane >> 4;
  const int col  = lane & 15;
  const int t0   = blockIdx.x * 128 + wave * 16;

  v16bf afr[8];
  const __bf16* kp = kin + ((size_t)n * S_DIM + t0 + col) * C_DIM + 16 * half;
#pragma unroll
  for (int kk = 0; kk < 8; ++kk) afr[kk] = load_frag_a(kp + kk * 32);

  float m_run[8], z_run[8];
#pragma unroll
  for (int r = 0; r < 8; ++r) { m_run[r] = -3.0e38f; z_run[r] = 0.0f; }

  const __bf16* qbase = qin + (size_t)n * S_DIM * C_DIM + 16 * half;
  for (int s0 = 0; s0 < S_DIM; s0 += 16) {
    v8f e = {};
    const __bf16* qp = qbase + (size_t)(s0 + col) * C_DIM;
#pragma unroll
    for (int kk = 0; kk < 8; ++kk) {
      v16bf b = load_frag_b(qp + kk * 32);
      e = __builtin_amdgcn_wmma_f32_16x16x32_bf16(
          false, afr[kk], false, b, (short)0, e, false, false);
    }
#pragma unroll
    for (int r = 0; r < 8; ++r) {
      float x = e[r];
      float mx = x;
      mx = fmaxf(mx, __shfl_xor(mx, 1, 32));
      mx = fmaxf(mx, __shfl_xor(mx, 2, 32));
      mx = fmaxf(mx, __shfl_xor(mx, 4, 32));
      mx = fmaxf(mx, __shfl_xor(mx, 8, 32));
      float mnew = fmaxf(m_run[r], mx);
      float p = __expf(x - mnew);
      p += __shfl_xor(p, 1, 32);
      p += __shfl_xor(p, 2, 32);
      p += __shfl_xor(p, 4, 32);
      p += __shfl_xor(p, 8, 32);
      z_run[r] = z_run[r] * __expf(m_run[r] - mnew) + p;
      m_run[r] = mnew;
    }
  }
  if (col == 0) {
#pragma unroll
    for (int r = 0; r < 8; ++r)
      Lout[(size_t)n * S_DIM + t0 + 8 * half + r] =
          m_run[r] + __logf(z_run[r]);
  }
}

// ---------------------------------------------------------------------------
// Kernel 3 (pass B): out[c,s] += sum_t v[c,t] * exp(e[t,s] - L[t]).
// K/V tiles (16 KB each) are identical across the 8 waves of a block ->
// TDM-stage them into double-buffered LDS; DMA of chunk i+1 overlaps compute
// of chunk i. Sync: wave0 s_wait_tensorcnt(0) -> barrier -> issue next.
// ---------------------------------------------------------------------------
__global__ __launch_bounds__(256) void attn_av_kernel(
    const __bf16* __restrict__ kin, const __bf16* __restrict__ qin,
    const __bf16* __restrict__ vin, const float* __restrict__ Lin,
    float* __restrict__ out)
{
  __shared__ __bf16 kT[2][32][256];   // (t, c) rows, 16 KB per buffer
  __shared__ __bf16 vT[2][256][32];   // (c, t) rows, 16 KB per buffer
  __shared__ __bf16 pT[8][16][32];    // per-wave P transpose (s, t)

  const int n     = blockIdx.y;
  const int wave  = threadIdx.x >> 5;
  const int lane  = threadIdx.x & 31;
  const int half  = lane >> 4;
  const int col   = lane & 15;
  const int sBase = blockIdx.x * 128 + wave * 16;

  // Preload Q B-fragments for this wave's s columns (whole C range)
  v16bf qfr[8];
  const __bf16* qp = qin + ((size_t)n * S_DIM + sBase + col) * C_DIM + 16 * half;
#pragma unroll
  for (int kk = 0; kk < 8; ++kk) qfr[kk] = load_frag_b(qp + kk * 32);

  v8f acc[16];
#pragma unroll
  for (int ct = 0; ct < 16; ++ct)
#pragma unroll
    for (int r = 0; r < 8; ++r) acc[ct][r] = 0.0f;

  const float*  Lp    = Lin + (size_t)n * S_DIM;
  const __bf16* kbase = kin + (size_t)n * S_DIM * C_DIM;       // (t, c)
  const __bf16* vbase = vin + (size_t)n * C_DIM * S_DIM;       // (c, t)

  // Prologue: DMA chunk 0 into buffer 0
  if (wave == 0) {
    tdm_load_2d_bf16((unsigned int)(uintptr_t)&kT[0][0][0], kbase,
                     C_DIM, 32, C_DIM, S_DIM, C_DIM);
    tdm_load_2d_bf16((unsigned int)(uintptr_t)&vT[0][0][0], vbase,
                     32, C_DIM, S_DIM, C_DIM, S_DIM);
  }

  for (int t0 = 0; t0 < S_DIM; t0 += 32) {
    const int cur = (t0 >> 5) & 1;
    if (wave == 0) __builtin_amdgcn_s_wait_tensorcnt(0);  // chunk t0 in LDS
    __syncthreads();                                      // ..visible to all
    if (wave == 0 && t0 + 32 < S_DIM) {                   // prefetch chunk t0+32
      const int nxt = cur ^ 1;
      tdm_load_2d_bf16((unsigned int)(uintptr_t)&kT[nxt][0][0],
                       kbase + (size_t)(t0 + 32) * C_DIM,
                       C_DIM, 32, C_DIM, S_DIM, C_DIM);
      tdm_load_2d_bf16((unsigned int)(uintptr_t)&vT[nxt][0][0],
                       vbase + (t0 + 32),
                       32, C_DIM, S_DIM, C_DIM, S_DIM);
    }

    // e tiles (2 x 16t x 16s), K from LDS, then p = exp(e - L) -> pT
#pragma unroll
    for (int j = 0; j < 2; ++j) {
      v8f e = {};
      const __bf16* kp = &kT[cur][16 * j + col][16 * half];
#pragma unroll
      for (int kk = 0; kk < 8; ++kk) {
        v16bf a = load_frag_a(kp + kk * 32);
        e = __builtin_amdgcn_wmma_f32_16x16x32_bf16(
            false, a, false, qfr[kk], (short)0, e, false, false);
      }
#pragma unroll
      for (int r = 0; r < 8; ++r) {
        float Lv = Lp[t0 + 16 * j + 8 * half + r];
        float p  = __expf(e[r] - Lv);
        pT[wave][col][16 * j + 8 * half + r] = (__bf16)p;  // transpose to (s,t)
      }
    }
    // Same-wave DS ops are in-order: build P B-fragment from LDS
    v16bf pb;
    {
      const __bf16* pp = &pT[wave][col][16 * half];
      v8bf lo = *(const v8bf*)pp;
      v8bf hi = *(const v8bf*)(pp + 8);
#pragma unroll
      for (int i = 0; i < 8; ++i) { pb[i] = lo[i]; pb[i + 8] = hi[i]; }
    }
    // V.P accumulation, V A-fragments from LDS
#pragma unroll
    for (int ct = 0; ct < 16; ++ct) {
      v16bf av = load_frag_a(&vT[cur][ct * 16 + col][16 * half]);
      acc[ct] = __builtin_amdgcn_wmma_f32_16x16x32_bf16(
          false, av, false, pb, (short)0, acc[ct], false, false);
    }
    __syncthreads();   // all waves done with buffer `cur` before it is re-DMA'd
  }

  // out already holds the Wl residual -> accumulate
#pragma unroll
  for (int ct = 0; ct < 16; ++ct)
#pragma unroll
    for (int r = 0; r < 8; ++r) {
      size_t idx = ((size_t)n * C_DIM + ct * 16 + r + 8 * half) * S_DIM +
                   sBase + col;
      out[idx] += acc[ct][r];
    }
}

// ---------------------------------------------------------------------------
extern "C" void kernel_launch(void* const* d_in, const int* in_sizes, int n_in,
                              void* d_out, int out_size, void* d_ws, size_t ws_size,
                              hipStream_t stream) {
  const float* xs = (const float*)d_in[0];
  const float* Wq = (const float*)d_in[1];
  const float* Wk = (const float*)d_in[2];
  const float* Wv = (const float*)d_in[3];
  const float* Wl = (const float*)d_in[4];
  float* out = (float*)d_out;

  const size_t QKV = (size_t)N_B * S_DIM * C_DIM;   // elements per tensor
  __bf16* q = (__bf16*)d_ws;                        // (n, S, C) bf16
  __bf16* k = q + QKV;                              // (n, S, C) bf16
  __bf16* v = k + QKV;                              // (n, C, S) bf16
  float*  L = (float*)(v + QKV);                    // (n, S) log-sum-exp

  proj_kernel<<<dim3(S_DIM / 16, N_B), 256, 0, stream>>>(
      xs, Wq, Wk, Wv, Wl, q, k, v, out);
  attn_stats_kernel<<<dim3(S_DIM / 128, N_B), 256, 0, stream>>>(k, q, L);
  attn_av_kernel<<<dim3(S_DIM / 128, N_B), 256, 0, stream>>>(k, q, v, L, out);
}
